// GCNContext_31035433681339
// MI455X (gfx1250) — matrix operand order repaired and checked by reading, not compiled, not run on hardware
//
#include <hip/hip_runtime.h>

#define NND 100000
#define NE  400000
#define KIN 1536
#define HID 768
#define LNEPS 1e-5f

typedef __attribute__((ext_vector_type(16))) __bf16 v16bf;
typedef __attribute__((ext_vector_type(8)))  __bf16 v8bf;
typedef __attribute__((ext_vector_type(8)))  float  v8f;

__device__ __forceinline__ unsigned short f2bf(float f) {
  union { float f; unsigned u; } c; c.f = f;
  unsigned u = c.u + 0x7FFFu + ((c.u >> 16) & 1u);   // round-to-nearest-even
  return (unsigned short)(u >> 16);
}
__device__ __forceinline__ float bf2f(unsigned short h) {
  union { unsigned u; float f; } c; c.u = ((unsigned)h) << 16;
  return c.f;
}
__device__ __forceinline__ float gelu_exact(float x) {
  return 0.5f * x * (1.0f + erff(x * 0.70710678118654752f));
}
__device__ __forceinline__ void wave_reduce2(float& s, float& sq) {
#pragma unroll
  for (int off = 16; off > 0; off >>= 1) {
    s  += __shfl_xor(s,  off, 32);
    sq += __shfl_xor(sq, off, 32);
  }
}

// ---------------- elementwise f32 -> bf16 (vec4) ----------------
__global__ void k_cvt_bf16(const float* __restrict__ in,
                           unsigned short* __restrict__ out, int n4) {
  int i = blockIdx.x * blockDim.x + threadIdx.x;
  if (i >= n4) return;
  float4 v = ((const float4*)in)[i];
  ushort4 o;
  o.x = f2bf(v.x); o.y = f2bf(v.y); o.z = f2bf(v.z); o.w = f2bf(v.w);
  ((ushort4*)out)[i] = o;
}

// ---------------- degree / rsqrt ----------------
__global__ void k_deg_init(float* __restrict__ deg) {
  int i = blockIdx.x * blockDim.x + threadIdx.x;
  if (i < NND) deg[i] = 1.0f;                       // self loop
}
__global__ void k_deg_edges(const long long* __restrict__ dst,
                            float* __restrict__ deg) {
  int e = blockIdx.x * blockDim.x + threadIdx.x;
  if (e < NE) atomicAdd(&deg[(int)dst[e]], 1.0f);
}
__global__ void k_deg_inv(float* __restrict__ deg) {
  int i = blockIdx.x * blockDim.x + threadIdx.x;
  if (i < NND) deg[i] = rsqrtf(deg[i]);             // deg >= 1 always
}

// ---------------- WMMA GEMM: C[M,768] = A[M,K](bf16) @ W[768,K](bf16)^T ----
// block = 256 threads (8 waves); block -> 32-row M-strip; wave w -> cols [96w,96w+96)
// Each wave computes 2 M-tiles x 6 N-tiles; B fragment shared by both M-tiles.
__global__ __launch_bounds__(256) void k_gemm_bf16(
    const unsigned short* __restrict__ A,   // [M,K] bf16 row-major
    const unsigned short* __restrict__ W,   // [768,K] bf16 row-major
    float* __restrict__ Cf,                 // f32 out (storeBF==0)
    unsigned short* __restrict__ Cb,        // bf16 out (storeBF==1)
    int K, int storeBF) {
  const int lane = threadIdx.x & 31;
  const int wave = threadIdx.x >> 5;
  const int half = lane >> 4;               // 0/1 lane-half per ISA layout
  const int l15  = lane & 15;
  const int row0 = blockIdx.x * 32;
  const int col0 = wave * 96;

  // A frag (16x32 bf16): lane half h holds K = h*16+{0..15} contiguous (matches B)
  const unsigned short* arow0 = A + (size_t)(row0 + l15) * K + half * 16;
  const unsigned short* arow1 = arow0 + (size_t)16 * K;
  // B frag (32x16 bf16): lane holds col n=l15, K = half*16+{0..15} contiguous
  const unsigned short* wbase = W + (size_t)(col0 + l15) * K + half * 16;

  v8f acc0[6], acc1[6];
#pragma unroll
  for (int t = 0; t < 6; ++t)
#pragma unroll
    for (int r = 0; r < 8; ++r) { acc0[t][r] = 0.0f; acc1[t][r] = 0.0f; }

  for (int k0 = 0; k0 < K; k0 += 32) {
    v8bf a0lo = *(const v8bf*)(arow0 + k0);
    v8bf a0hi = *(const v8bf*)(arow0 + k0 + 8);
    v8bf a1lo = *(const v8bf*)(arow1 + k0);
    v8bf a1hi = *(const v8bf*)(arow1 + k0 + 8);
    v16bf a0, a1;
#pragma unroll
    for (int i = 0; i < 8; ++i) {
      a0[i] = a0lo[i]; a0[8 + i] = a0hi[i];
      a1[i] = a1lo[i]; a1[8 + i] = a1hi[i];
    }
#pragma unroll
    for (int t = 0; t < 6; ++t) {
      v16bf b = *(const v16bf*)(wbase + (size_t)t * 16 * K + k0);
      acc0[t] = __builtin_amdgcn_wmma_f32_16x16x32_bf16(
          false, a0, false, b, (short)0, acc0[t], false, false);
      acc1[t] = __builtin_amdgcn_wmma_f32_16x16x32_bf16(
          false, a1, false, b, (short)0, acc1[t], false, false);
    }
  }

  // C/D layout: lane l, reg r -> (m = r + 8*half, n = l15) within tile
  const int mbase = row0 + 8 * half;
#pragma unroll
  for (int t = 0; t < 6; ++t) {
    const int col = col0 + t * 16 + l15;
#pragma unroll
    for (int r = 0; r < 8; ++r) {
      size_t idx0 = (size_t)(mbase + r) * HID + col;
      size_t idx1 = (size_t)(mbase + 16 + r) * HID + col;
      if (storeBF) {
        Cb[idx0] = f2bf(acc0[t][r]);
        Cb[idx1] = f2bf(acc1[t][r]);
      } else {
        Cf[idx0] = acc0[t][r];
        Cf[idx1] = acc1[t][r];
      }
    }
  }
}

// ---------------- LayerNorm after proj: h = LN(t + proj_b) ----------------
__global__ __launch_bounds__(256) void k_ln0(
    const float* __restrict__ tin, const float* __restrict__ pb,
    const float* __restrict__ g, const float* __restrict__ bta,
    float* __restrict__ hf, unsigned short* __restrict__ hb) {
  const int lane = threadIdx.x & 31, wave = threadIdx.x >> 5;
  const int row = blockIdx.x * 8 + wave;          // 100000 = 12500*8 exact
  const size_t base = (size_t)row * HID;
  float v[24];
  float s = 0.f, sq = 0.f;
#pragma unroll
  for (int j = 0; j < 6; ++j) {
    const int col = j * 128 + lane * 4;
    float4 t  = *(const float4*)(tin + base + col);
    float4 pv = *(const float4*)(pb + col);
    float x0 = t.x + pv.x, x1 = t.y + pv.y, x2 = t.z + pv.z, x3 = t.w + pv.w;
    v[4*j+0]=x0; v[4*j+1]=x1; v[4*j+2]=x2; v[4*j+3]=x3;
    s  += (x0 + x1) + (x2 + x3);
    sq += (x0*x0 + x1*x1) + (x2*x2 + x3*x3);
  }
  wave_reduce2(s, sq);
  const float mean = s * (1.0f / HID);
  const float rstd = rsqrtf(sq * (1.0f / HID) - mean * mean + LNEPS);
#pragma unroll
  for (int j = 0; j < 6; ++j) {
    const int col = j * 128 + lane * 4;
    float4 gv = *(const float4*)(g + col);
    float4 bv = *(const float4*)(bta + col);
    float4 o;
    o.x = (v[4*j+0] - mean) * rstd * gv.x + bv.x;
    o.y = (v[4*j+1] - mean) * rstd * gv.y + bv.y;
    o.z = (v[4*j+2] - mean) * rstd * gv.z + bv.z;
    o.w = (v[4*j+3] - mean) * rstd * gv.w + bv.w;
    *(float4*)(hf + base + col) = o;
    ushort4 ob; ob.x = f2bf(o.x); ob.y = f2bf(o.y); ob.z = f2bf(o.z); ob.w = f2bf(o.w);
    *(ushort4*)(hb + base + col) = ob;
  }
}

// ---------------- self-loop init: agg[v] = dinv[v]^2 * t[v] ----------------
__global__ void k_selfloop(const unsigned short* __restrict__ t,
                           const float* __restrict__ dinv,
                           float* __restrict__ agg) {
  int i = blockIdx.x * blockDim.x + threadIdx.x;     // over N*HID/4
  if (i >= NND * (HID / 4)) return;
  int row = i / (HID / 4);
  float w = dinv[row]; w = w * w;
  ushort4 tv = ((const ushort4*)t)[i];
  float4 o = make_float4(w * bf2f(tv.x), w * bf2f(tv.y),
                         w * bf2f(tv.z), w * bf2f(tv.w));
  ((float4*)agg)[i] = o;
}

// ---------------- edge scatter: agg[dst] += dinv[s]*dinv[d]*t[src] --------
__global__ __launch_bounds__(256) void k_edge_scatter(
    const long long* __restrict__ src, const long long* __restrict__ dst,
    const float* __restrict__ dinv, const unsigned short* __restrict__ t,
    float* __restrict__ agg) {
  const int lane = threadIdx.x & 31, wave = threadIdx.x >> 5;
  const int e = blockIdx.x * 8 + wave;               // 400000 = 50000*8 exact
  if (e >= NE) return;
  const int s = (int)src[e], d = (int)dst[e];
  const float w = dinv[s] * dinv[d];
  const unsigned short* trow = t + (size_t)s * HID;
  float* arow = agg + (size_t)d * HID;
#pragma unroll
  for (int j = 0; j < 6; ++j) {
    const int col = j * 128 + lane * 4;
    ushort4 tv = *(const ushort4*)(trow + col);
    atomicAdd(arow + col + 0, w * bf2f(tv.x));
    atomicAdd(arow + col + 1, w * bf2f(tv.y));
    atomicAdd(arow + col + 2, w * bf2f(tv.z));
    atomicAdd(arow + col + 3, w * bf2f(tv.w));
  }
}

// ---------------- hop epilogue: h = LN(h + gelu(agg + gcn_b)) -------------
__global__ __launch_bounds__(256) void k_hopln(
    const float* __restrict__ agg, const float* __restrict__ gb,
    const float* __restrict__ hin,
    const float* __restrict__ g, const float* __restrict__ bta,
    float* __restrict__ hf, unsigned short* __restrict__ hb) {
  const int lane = threadIdx.x & 31, wave = threadIdx.x >> 5;
  const int row = blockIdx.x * 8 + wave;
  const size_t base = (size_t)row * HID;
  float v[24];
  float s = 0.f, sq = 0.f;
#pragma unroll
  for (int j = 0; j < 6; ++j) {
    const int col = j * 128 + lane * 4;
    float4 a  = *(const float4*)(agg + base + col);
    float4 bb = *(const float4*)(gb + col);
    float4 h  = *(const float4*)(hin + base + col);
    float x0 = h.x + gelu_exact(a.x + bb.x);
    float x1 = h.y + gelu_exact(a.y + bb.y);
    float x2 = h.z + gelu_exact(a.z + bb.z);
    float x3 = h.w + gelu_exact(a.w + bb.w);
    v[4*j+0]=x0; v[4*j+1]=x1; v[4*j+2]=x2; v[4*j+3]=x3;
    s  += (x0 + x1) + (x2 + x3);
    sq += (x0*x0 + x1*x1) + (x2*x2 + x3*x3);
  }
  wave_reduce2(s, sq);
  const float mean = s * (1.0f / HID);
  const float rstd = rsqrtf(sq * (1.0f / HID) - mean * mean + LNEPS);
#pragma unroll
  for (int j = 0; j < 6; ++j) {
    const int col = j * 128 + lane * 4;
    float4 gv = *(const float4*)(g + col);
    float4 bv = *(const float4*)(bta + col);
    float4 o;
    o.x = (v[4*j+0] - mean) * rstd * gv.x + bv.x;
    o.y = (v[4*j+1] - mean) * rstd * gv.y + bv.y;
    o.z = (v[4*j+2] - mean) * rstd * gv.z + bv.z;
    o.w = (v[4*j+3] - mean) * rstd * gv.w + bv.w;
    *(float4*)(hf + base + col) = o;
    ushort4 ob; ob.x = f2bf(o.x); ob.y = f2bf(o.y); ob.z = f2bf(o.z); ob.w = f2bf(o.w);
    *(ushort4*)(hb + base + col) = ob;
  }
}

extern "C" void kernel_launch(void* const* d_in, const int* in_sizes, int n_in,
                              void* d_out, int out_size, void* d_ws, size_t ws_size,
                              hipStream_t stream) {
  (void)in_sizes; (void)n_in; (void)out_size; (void)ws_size;
  const float*     x      = (const float*)d_in[0];
  const long long* ei     = (const long long*)d_in[1];   // [2,E] int64
  const float*     proj_w = (const float*)d_in[2];
  const float*     proj_b = (const float*)d_in[3];
  const float*     ln0_g  = (const float*)d_in[4];
  const float*     ln0_b  = (const float*)d_in[5];
  const float*     gcn_w  = (const float*)d_in[6];
  const float*     gcn_b  = (const float*)d_in[7];
  const float*     lnh_g  = (const float*)d_in[8];
  const float*     lnh_b  = (const float*)d_in[9];
  float* out = (float*)d_out;

  char* ws = (char*)d_ws;
  size_t off = 0;
  auto take = [&](size_t bytes) {
    void* p = ws + off;
    off = (off + bytes + 255) & ~(size_t)255;
    return p;
  };
  float*          dinv = (float*)take((size_t)NND * 4);
  float*          hf   = (float*)take((size_t)NND * HID * 4);
  unsigned short* hb   = (unsigned short*)take((size_t)NND * HID * 2);
  unsigned short* xbf  = (unsigned short*)take((size_t)NND * KIN * 2); // x bf16, reused as t bf16
  float*          agg  = (float*)take((size_t)NND * HID * 4);          // proj-out, then agg
  unsigned short* pwb  = (unsigned short*)take((size_t)HID * KIN * 2);
  unsigned short* gwb  = (unsigned short*)take((size_t)3 * HID * HID * 2);
  unsigned short* tbf  = xbf;

  // --- convert to bf16 ---
  k_cvt_bf16<<<(NND * (KIN/4) + 255) / 256, 256, 0, stream>>>(x, xbf, NND * (KIN/4));
  k_cvt_bf16<<<(HID * (KIN/4) + 255) / 256, 256, 0, stream>>>(proj_w, pwb, HID * (KIN/4));
  k_cvt_bf16<<<(3 * HID * (HID/4) + 255) / 256, 256, 0, stream>>>(gcn_w, gwb, 3 * HID * (HID/4));

  // --- degree -> dinv ---
  k_deg_init<<<(NND + 255) / 256, 256, 0, stream>>>(dinv);
  k_deg_edges<<<(NE + 255) / 256, 256, 0, stream>>>(ei + NE, dinv);
  k_deg_inv<<<(NND + 255) / 256, 256, 0, stream>>>(dinv);

  // --- proj GEMM + LN ---
  k_gemm_bf16<<<NND / 32, 256, 0, stream>>>(xbf, pwb, agg, (unsigned short*)nullptr, KIN, 0);
  k_ln0<<<NND / 8, 256, 0, stream>>>(agg, proj_b, ln0_g, ln0_b, hf, hb);

  // --- 3 GCN hops ---
  for (int i = 0; i < 3; ++i) {
    k_gemm_bf16<<<NND / 32, 256, 0, stream>>>(hb, gwb + (size_t)i * HID * HID,
                                              (float*)nullptr, tbf, HID, 1);
    k_selfloop<<<(NND * (HID/4) + 255) / 256, 256, 0, stream>>>(tbf, dinv, agg);
    k_edge_scatter<<<NE / 8, 256, 0, stream>>>(ei, ei + NE, dinv, tbf, agg);
    k_hopln<<<NND / 8, 256, 0, stream>>>(agg, gcn_b + (size_t)i * HID, hf,
                                         lnh_g + (size_t)i * HID, lnh_b + (size_t)i * HID,
                                         (i == 2) ? out : hf, hb);
  }
}
